// Transformer_5626407157925
// MI455X (gfx1250) — matrix operand order repaired and checked by reading, not compiled, not run on hardware
//
#include <hip/hip_runtime.h>
#include <math.h>

// Problem constants (from the reference)
#define B_   256
#define L_   100
#define F_   8
#define P_   4
#define E_   64
#define H_   4
#define D_   64
#define C_   32
#define V_   10000
#define HD_  256
#define TIN_ 1312   // F*E + F*E + P*E + C

typedef __attribute__((ext_vector_type(16))) __bf16 v16bf;
typedef __attribute__((ext_vector_type(8)))  float  v8f;
typedef __attribute__((ext_vector_type(4)))  int    v4i;

// gfx1250 async memory->LDS path (ASYNCcnt tracked), if the toolchain has it.
#if defined(__has_builtin)
#if __has_builtin(__builtin_amdgcn_global_load_async_to_lds_b128) && \
    __has_builtin(__builtin_amdgcn_s_wait_asynccnt)
#define HAVE_ASYNC_LDS 1
#endif
#endif

static __device__ __forceinline__ unsigned short bfbits(__bf16 x) {
  union { __bf16 b; unsigned short s; } u; u.b = x; return u.s;
}

static __device__ __forceinline__ v8f wmma_bf16(v16bf a, v16bf b, v8f c) {
  // 8 args: (neg_a, A, neg_b, B, c_mod, C, reuse_a, reuse_b)
  return __builtin_amdgcn_wmma_f32_16x16x32_bf16(false, a, false, b, (short)0, c,
                                                 false, false);
}

// --- A fragment (16x32 bf16, M x K) from a row-major bf16 LDS tile.
// ISA layout: lanes 0-15 hold M=lane, K=0..7 (v0..3) and K=16..23 (v4..7);
// lanes 16-31 hold M=lane-16, K=8..15 (v0..3) and K=24..31 (v4..7).
// strideElems must be even; p must be dword aligned.
static __device__ __forceinline__ v16bf load_a_frag(const __bf16* p, int strideElems) {
  int lane = threadIdx.x & 31;
  int m  = lane & 15;
  int d0 = (lane & 16) ? 4 : 0;    // dword index of first K pair group
  int d1 = (lane & 16) ? 12 : 8;   // dword index of second K pair group
  union { v16bf v; unsigned u[8]; } r;
  const unsigned* row = (const unsigned*)(p + (size_t)m * strideElems);
#pragma unroll
  for (int j = 0; j < 4; ++j) r.u[j]     = row[d0 + j];
#pragma unroll
  for (int j = 0; j < 4; ++j) r.u[4 + j] = row[d1 + j];
  return r.v;
}

// --- B fragment (32x16 bf16, K x N) from the pre-packed weight image.
// Packed so that each lane's 8 dwords are contiguous: tile base + lane*8 + j.
static __device__ __forceinline__ v16bf load_b_frag(const unsigned* tile) {
  int lane = threadIdx.x & 31;
  union { v16bf v; unsigned u[8]; } r;
  const unsigned* q = tile + lane * 8;
#pragma unroll
  for (int j = 0; j < 8; ++j) r.u[j] = q[j];
  return r.v;
}

// --- C/D (16x16 f32) store: lanes 0-15 -> M=r, N=lane; lanes 16-31 -> M=8+r.
static __device__ __forceinline__ void store_c_f32(float* dst, int stride, int n0, v8f c) {
  int lane = threadIdx.x & 31;
  int n  = n0 + (lane & 15);
  int mb = (lane & 16) ? 8 : 0;
#pragma unroll
  for (int r = 0; r < 8; ++r) dst[(size_t)(mb + r) * stride + n] = c[r];
}

// =====================================================================
// Pack a row-major (K x N) fp32 matrix (nmat of them, contiguous) into
// the WMMA bf16 B-fragment image: dword id = ((m*KT*NT + kt*NT + nt)*32
// + lane)*8 + j holds {hi: w[k+1,n], lo: w[k,n]} with
// n = nt*16 + (lane&15), k = kt*32 + (lane&16 ? 8 : 0) + 2*j + (j >= 4 ? 8 : 0).
// =====================================================================
__global__ __launch_bounds__(256) void pack_b_kernel(
    const float* __restrict__ src, unsigned* __restrict__ dst,
    int K, int N, int nmat)
{
  int KT = K >> 5, NT = N >> 4;
  long perMat = (long)KT * NT * 256;
  long total  = perMat * nmat;
  long id = (long)blockIdx.x * 256 + threadIdx.x;
  if (id >= total) return;
  int  m      = (int)(id / perMat);
  long rem    = id - (long)m * perMat;
  int  tile   = (int)(rem >> 8);
  int  within = (int)(rem & 255);
  int  lane = within >> 3, j = within & 7;
  int  kt = tile / NT, nt = tile % NT;
  int  n = nt * 16 + (lane & 15);
  int  k = kt * 32 + ((lane & 16) ? 8 : 0) + 2 * j + ((j >= 4) ? 8 : 0);
  const float* sm = src + (long)m * K * N;
  __bf16 lo = (__bf16)sm[(long)k * N + n];
  __bf16 hi = (__bf16)sm[(long)(k + 1) * N + n];
  dst[id] = ((unsigned)bfbits(hi) << 16) | (unsigned)bfbits(lo);
}

// =====================================================================
// Fused gather + QKV + attention + lin + LayerNorm, accumulated over L
// via f32 atomics.  grid = (L, B/2), block = 128 (4 wave32).
// Rows 0..7 = batch pair*2 features, rows 8..15 = batch pair*2+1.
// =====================================================================
__global__ __launch_bounds__(128) void attn_kernel(
    const int*   __restrict__ ubs_feature,   // (B,L,F)
    const float* __restrict__ item_tables,   // (F,V,E)
    const unsigned* __restrict__ qpk,        // packed (L, KT=2, NT=16)
    const unsigned* __restrict__ kpk,
    const unsigned* __restrict__ vpk,
    const unsigned* __restrict__ lpk,        // packed (L, KT=8, NT=4)
    const float* __restrict__ lin_b,         // (L,E)
    const float* __restrict__ ln_g,          // (E)
    const float* __restrict__ ln_bv,         // (E)
    float*       __restrict__ acc)           // (B,F,E) f32 accumulator
{
  // Manual LDS arena (62 KB total) with barrier-separated overlap:
  //   [0,    8K)  : ue_f32 staging (4 KB, early)  /  att_bf (8 KB, late)
  //   [8K,  10K)  : ue_s   bf16 16x64
  //   [10K, 58K)  : qs, ks, vsh  f32 16x256 each
  //   [58K, 62K)  : lino   f32 16x64
  __shared__ __align__(16) unsigned char smem[8192 + 2048 + 3 * 16384 + 4096];
  float*  ue_f32 = (float*)smem;                        // overlaps att_bf
  __bf16* att_bf = (__bf16*)smem;
  __bf16* ue_s   = (__bf16*)(smem + 8192);
  float*  qs     = (float*)(smem + 8192 + 2048);
  float*  ks     = (float*)(smem + 8192 + 2048 + 16384);
  float*  vsh    = (float*)(smem + 8192 + 2048 + 2 * 16384);
  float*  lino   = (float*)(smem + 8192 + 2048 + 3 * 16384);

  const int l    = blockIdx.x;
  const int pair = blockIdx.y;
  const int t    = threadIdx.x;
  const int wave = t >> 5;

  const unsigned* lb = lpk + (size_t)l * (8 * 4 * 256);

  // ---- gather ue rows (2 batches x 8 features x 64 f32) into LDS
#ifdef HAVE_ASYNC_LDS
  {
    // 128 threads; 8 threads per row, each copies 32 B = 2 x b128 async.
    int m  = t >> 3;           // row 0..15
    int q4 = t & 7;            // 8-float chunk within the row
    int bb = m >> 3, f = m & 7;
    int b  = pair * 2 + bb;
    int id = ubs_feature[(b * L_ + l) * F_ + f];
    const float* src = item_tables + ((size_t)f * V_ + id) * E_ + q4 * 8;
    float*       dst = ue_f32 + m * 64 + q4 * 8;
    __builtin_amdgcn_global_load_async_to_lds_b128(
        (__attribute__((address_space(1))) v4i*)src,
        (__attribute__((address_space(3))) v4i*)dst, 0, 0);
    __builtin_amdgcn_global_load_async_to_lds_b128(
        (__attribute__((address_space(1))) v4i*)(src + 4),
        (__attribute__((address_space(3))) v4i*)(dst + 4), 0, 0);
    __builtin_amdgcn_s_wait_asynccnt(0);
  }
  __syncthreads();
  for (int idx = t; idx < 16 * 64; idx += 128)
    ue_s[idx] = (__bf16)ue_f32[idx];
#else
  for (int idx = t; idx < 16 * 64; idx += 128) {
    int m = idx >> 6, e = idx & 63;
    int bb = m >> 3, f = m & 7;
    int b  = pair * 2 + bb;
    int id = ubs_feature[(b * L_ + l) * F_ + f];
    ue_s[idx] = (__bf16)item_tables[((size_t)f * V_ + id) * E_ + e];
  }
#endif

  // Straight-line prefetch of this layer's lin_w packed image (32 KB) into
  // near caches while the QKV stage runs: 128 threads x 256 B slices.
  __builtin_prefetch(lb + (size_t)t * 64, 0, 3);

  __syncthreads();

  // ---- QKV projections: M=16, K=64 (2 k-steps), N=256 (16 n-tiles / 4 waves)
  {
    v16bf a0 = load_a_frag(ue_s, 64);
    v16bf a1 = load_a_frag(ue_s + 32, 64);
    const unsigned* qb = qpk + (size_t)l * (2 * 16 * 256);
    const unsigned* kb = kpk + (size_t)l * (2 * 16 * 256);
    const unsigned* vb = vpk + (size_t)l * (2 * 16 * 256);
    for (int nt = wave; nt < 16; nt += 4) {
      v8f cq = {}, ck = {}, cv = {};
      cq = wmma_bf16(a0, load_b_frag(qb + (size_t)(0 * 16 + nt) * 256), cq);
      cq = wmma_bf16(a1, load_b_frag(qb + (size_t)(1 * 16 + nt) * 256), cq);
      ck = wmma_bf16(a0, load_b_frag(kb + (size_t)(0 * 16 + nt) * 256), ck);
      ck = wmma_bf16(a1, load_b_frag(kb + (size_t)(1 * 16 + nt) * 256), ck);
      cv = wmma_bf16(a0, load_b_frag(vb + (size_t)(0 * 16 + nt) * 256), cv);
      cv = wmma_bf16(a1, load_b_frag(vb + (size_t)(1 * 16 + nt) * 256), cv);
      store_c_f32(qs,  HD_, nt * 16, cq);
      store_c_f32(ks,  HD_, nt * 16, ck);
      store_c_f32(vsh, HD_, nt * 16, cv);
    }
  }
  __syncthreads();

  // ---- attention: per (batch-in-pair, head, feature-row): 8x8 softmax
  if (t < 64) {
    int f = t & 7, h = (t >> 3) & 3, bb = (t >> 5) & 1;
    int row = bb * 8 + f;
    const float* qrow = qs + row * HD_ + h * D_;
    float sc[8];
    float mx = -3.0e30f;
#pragma unroll
    for (int g = 0; g < 8; ++g) {
      const float* krow = ks + (bb * 8 + g) * HD_ + h * D_;
      float s = 0.f;
      for (int d = 0; d < D_; ++d) s += qrow[d] * krow[d];
      s *= 0.125f;                 // 1/sqrt(D)
      sc[g] = s;
      mx = fmaxf(mx, s);
    }
    float se = 0.f;
#pragma unroll
    for (int g = 0; g < 8; ++g) { sc[g] = expf(sc[g] - mx); se += sc[g]; }
    float inv = 1.f / se;
    for (int d = 0; d < D_; ++d) {
      float a = 0.f;
#pragma unroll
      for (int g = 0; g < 8; ++g) a += sc[g] * vsh[(bb * 8 + g) * HD_ + h * D_ + d];
      att_bf[row * HD_ + h * D_ + d] = (__bf16)(a * inv);
    }
  }
  __syncthreads();

  // ---- lin projection: M=16, K=256 (8 k-steps), N=64 (1 n-tile per wave)
  {
    for (int nt = wave; nt < 4; nt += 4) {
      v8f c = {};
      for (int kt = 0; kt < 8; ++kt) {
        v16bf a = load_a_frag(att_bf + kt * 32, HD_);
        c = wmma_bf16(a, load_b_frag(lb + (size_t)(kt * 4 + nt) * 256), c);
      }
      store_c_f32(lino, 64, nt * 16, c);
    }
  }
  __syncthreads();

  // ---- per-row LayerNorm(att_out + lin_b), atomic-accumulate over L
  if (t < 16) {
    int bb = t >> 3, f = t & 7;
    int b  = pair * 2 + bb;
    const float* lbv = lin_b + l * E_;
    float mean = 0.f;
    for (int e = 0; e < E_; ++e) mean += lino[t * E_ + e] + lbv[e];
    mean *= (1.f / E_);
    float var = 0.f;
    for (int e = 0; e < E_; ++e) {
      float d = lino[t * E_ + e] + lbv[e] - mean;
      var += d * d;
    }
    var *= (1.f / E_);
    float rs = rsqrtf(var + 1e-3f);
    float* dst = acc + ((size_t)b * F_ + f) * E_;
    for (int e = 0; e < E_; ++e) {
      float y = (lino[t * E_ + e] + lbv[e] - mean) * rs * ln_g[e] + ln_bv[e];
      atomicAdd(dst + e, y);
    }
  }
}

// =====================================================================
// Tower: generic M=16 WMMA GEMM layer (bf16 in LDS -> +bias -> bf16 LDS)
// =====================================================================
static __device__ void mm_layer(const __bf16* inb, int K,
                                const unsigned* wpk, int N,
                                const float* bias, __bf16* outb)
{
  int wave = threadIdx.x >> 5;
  int NT = N >> 4, KT = K >> 5;
  for (int nt = wave; nt < NT; nt += 4) {
    v8f c = {};
    for (int kt = 0; kt < KT; ++kt) {
      v16bf a = load_a_frag(inb + kt * 32, K);
      v16bf b = load_b_frag(wpk + (size_t)(kt * NT + nt) * 256);
      c = wmma_bf16(a, b, c);
    }
    int lane = threadIdx.x & 31;
    int n  = nt * 16 + (lane & 15);
    int mb = (lane & 16) ? 8 : 0;
    float bv = bias[n];
#pragma unroll
    for (int r = 0; r < 8; ++r)
      outb[(size_t)(mb + r) * N + n] = (__bf16)(c[r] + bv);
  }
}

static __device__ void ln_relu(const __bf16* h, int W,
                               const float* g, const float* b, __bf16* out)
{
  int t = threadIdx.x;
  if (t < 16) {
    float mean = 0.f;
    for (int j = 0; j < W; ++j) mean += (float)h[(size_t)t * W + j];
    mean /= (float)W;
    float var = 0.f;
    for (int j = 0; j < W; ++j) {
      float d = (float)h[(size_t)t * W + j] - mean;
      var += d * d;
    }
    var /= (float)W;
    float rs = rsqrtf(var + 1e-3f);
    for (int j = 0; j < W; ++j) {
      float y = ((float)h[(size_t)t * W + j] - mean) * rs * g[j] + b[j];
      out[(size_t)t * W + j] = (__bf16)fmaxf(y, 0.f);
    }
  }
}

// grid = B/16 blocks, 128 threads
__global__ __launch_bounds__(128) void tower_kernel(
    const int*   __restrict__ target_ad,       // (B,F)
    const int*   __restrict__ profile_feature, // (B,P)
    const float* __restrict__ item_tables,     // (F,V,E)
    const float* __restrict__ profile_tables,  // (P,V,E)
    const float* __restrict__ ctx,             // (B,C)
    const float* __restrict__ acc,             // (B,F*E) LN-sum over L
    const unsigned* __restrict__ w1pk, const unsigned* __restrict__ w2pk,
    const unsigned* __restrict__ w3pk,
    const float* __restrict__ b1, const float* __restrict__ g1, const float* __restrict__ bb1,
    const float* __restrict__ b2, const float* __restrict__ g2, const float* __restrict__ bb2,
    const float* __restrict__ b3, const float* __restrict__ g3, const float* __restrict__ bb3,
    const float* __restrict__ w4, const float* __restrict__ b4,
    float* __restrict__ out)
{
  __shared__ __align__(16) __bf16 s_in[16 * TIN_];  // 42 KB; reused for LN outs
  __shared__ __align__(16) __bf16 s_h [16 * 1024];  // 32 KB; matmul(+bias) outs

  int t  = threadIdx.x;
  int b0 = blockIdx.x * 16;

  // ---- build concat input [relu(acc/L) | te | pe | ctx] as bf16
  for (int idx = t; idx < 16 * TIN_; idx += 128) {
    int m = idx / TIN_, j = idx - m * TIN_;
    int b = b0 + m;
    float v;
    if (j < 512) {
      v = fmaxf(acc[(size_t)b * 512 + j] * 0.01f, 0.f);   // /L, relu
    } else if (j < 1024) {
      int f = (j - 512) >> 6, e = j & 63;
      int id = target_ad[b * F_ + f];
      v = item_tables[((size_t)f * V_ + id) * E_ + e];
    } else if (j < 1280) {
      int p = (j - 1024) >> 6, e = j & 63;
      int id = profile_feature[b * P_ + p];
      v = profile_tables[((size_t)p * V_ + id) * E_ + e];
    } else {
      v = ctx[b * C_ + (j - 1280)];
    }
    s_in[(size_t)m * TIN_ + j] = (__bf16)v;
  }
  __syncthreads();

  mm_layer(s_in, TIN_, w1pk, 1024, b1, s_h);   __syncthreads();
  ln_relu(s_h, 1024, g1, bb1, s_in);           __syncthreads();
  mm_layer(s_in, 1024, w2pk,  512, b2, s_h);   __syncthreads();
  ln_relu(s_h,  512, g2, bb2, s_in);           __syncthreads();
  mm_layer(s_in,  512, w3pk,  256, b3, s_h);   __syncthreads();
  ln_relu(s_h,  256, g3, bb3, s_in);           __syncthreads();

  // ---- final 256 -> 1 + sigmoid
  if (t < 16) {
    float a = b4[0];
    for (int k = 0; k < 256; ++k)
      a += (float)s_in[(size_t)t * 256 + k] * w4[k];
    out[b0 + t] = 1.f / (1.f + expf(-a));
  }
}

// =====================================================================
// Host side
// =====================================================================
extern "C" void kernel_launch(void* const* d_in, const int* in_sizes, int n_in,
                              void* d_out, int out_size, void* d_ws, size_t ws_size,
                              hipStream_t stream)
{
  (void)in_sizes; (void)n_in; (void)out_size; (void)ws_size;

  const int*   target_ad = (const int*)d_in[0];
  const int*   ubs_feat  = (const int*)d_in[1];
  const int*   prof_feat = (const int*)d_in[2];
  const float* ctx       = (const float*)d_in[3];
  const float* item_tab  = (const float*)d_in[4];
  const float* prof_tab  = (const float*)d_in[5];
  const float* q_w   = (const float*)d_in[6];
  const float* k_w   = (const float*)d_in[7];
  const float* v_w   = (const float*)d_in[8];
  const float* lin_w = (const float*)d_in[9];
  const float* lin_b = (const float*)d_in[10];
  const float* ln_g  = (const float*)d_in[11];
  const float* ln_b  = (const float*)d_in[12];
  const float* t_w1  = (const float*)d_in[13];
  const float* t_b1  = (const float*)d_in[14];
  const float* t_g1  = (const float*)d_in[15];
  const float* t_bb1 = (const float*)d_in[16];
  const float* t_w2  = (const float*)d_in[17];
  const float* t_b2  = (const float*)d_in[18];
  const float* t_g2  = (const float*)d_in[19];
  const float* t_bb2 = (const float*)d_in[20];
  const float* t_w3  = (const float*)d_in[21];
  const float* t_b3  = (const float*)d_in[22];
  const float* t_g3  = (const float*)d_in[23];
  const float* t_bb3 = (const float*)d_in[24];
  const float* t_w4  = (const float*)d_in[25];
  const float* t_b4  = (const float*)d_in[26];
  float* out = (float*)d_out;

  // Workspace layout (dwords). Total ~17.6 MB.
  const size_t SZ_QKV = (size_t)L_ * 2 * 16 * 256;   // 819200
  const size_t SZ_LIN = (size_t)L_ * 8 *  4 * 256;   // 819200
  const size_t SZ_W1  = (size_t)41 * 64 * 256;       // 671744
  const size_t SZ_W2  = (size_t)32 * 32 * 256;       // 262144
  const size_t SZ_W3  = (size_t)16 * 16 * 256;       //  65536
  unsigned* ws   = (unsigned*)d_ws;
  unsigned* qpk  = ws;
  unsigned* kpk  = qpk + SZ_QKV;
  unsigned* vpk  = kpk + SZ_QKV;
  unsigned* lpk  = vpk + SZ_QKV;
  unsigned* w1pk = lpk + SZ_LIN;
  unsigned* w2pk = w1pk + SZ_W1;
  unsigned* w3pk = w2pk + SZ_W2;
  float*    accb = (float*)(w3pk + SZ_W3);           // (B, F*E) f32

  auto launch_pack = [&](const float* src, unsigned* dst, int K, int N, int nmat) {
    long total = (long)(K >> 5) * (N >> 4) * 256 * nmat;
    unsigned blocks = (unsigned)((total + 255) / 256);
    pack_b_kernel<<<blocks, 256, 0, stream>>>(src, dst, K, N, nmat);
  };
  launch_pack(q_w,   qpk,  64,  256, L_);
  launch_pack(k_w,   kpk,  64,  256, L_);
  launch_pack(v_w,   vpk,  64,  256, L_);
  launch_pack(lin_w, lpk, 256,   64, L_);
  launch_pack(t_w1, w1pk, TIN_, 1024, 1);
  launch_pack(t_w2, w2pk, 1024,  512, 1);
  launch_pack(t_w3, w3pk,  512,  256, 1);

  (void)hipMemsetAsync(accb, 0, (size_t)B_ * F_ * E_ * sizeof(float), stream);

  attn_kernel<<<dim3(L_, B_ / 2), 128, 0, stream>>>(
      ubs_feat, item_tab, qpk, kpk, vpk, lpk, lin_b, ln_g, ln_b, accb);

  tower_kernel<<<B_ / 16, 128, 0, stream>>>(
      target_ad, prof_feat, item_tab, prof_tab, ctx, accb,
      w1pk, w2pk, w3pk,
      t_b1, t_g1, t_bb1, t_b2, t_g2, t_bb2, t_b3, t_g3, t_bb3,
      t_w4, t_b4, out);
}
